// HiGLDP_3075196584313
// MI455X (gfx1250) — compile-verified, matrix operand order checked by reading, not run on hardware
//
#include <hip/hip_runtime.h>
#include <hip/hip_bf16.h>
#include <math.h>

typedef unsigned short u16;
typedef unsigned int u32;
typedef __attribute__((ext_vector_type(16))) __bf16 v16bf;
typedef __attribute__((ext_vector_type(8)))  float  v8f;

// ---------------------------------------------------------------------------
// elementwise helpers
// ---------------------------------------------------------------------------
__global__ void k_cast_bf16(const float* __restrict__ src, u16* __restrict__ dst, int n) {
    int i = blockIdx.x * blockDim.x + threadIdx.x;
    if (i < n) {
        u32 b = __float_as_uint(src[i]);
        u32 r = (b + 0x7FFFu + ((b >> 16) & 1u)) >> 16;   // RNE f32->bf16
        dst[i] = (u16)r;
    }
}

__global__ void k_fill_f32(float* __restrict__ p, float v, int n) {
    int i = blockIdx.x * blockDim.x + threadIdx.x;
    if (i < n) p[i] = v;
}

__global__ void k_relu(float* __restrict__ p, int n) {
    int i = blockIdx.x * blockDim.x + threadIdx.x;
    if (i < n) p[i] = fmaxf(p[i], 0.0f);
}

__global__ void k_rsqrt(float* __restrict__ p, int n) {
    int i = blockIdx.x * blockDim.x + threadIdx.x;
    if (i < n) p[i] = rsqrtf(p[i]);
}

// ---------------------------------------------------------------------------
// WMMA bf16 GEMM:  C[M,N] = act(A[M,K] @ B[K,N] + bias)
// 256 thr = 8 waves (4M x 2N), wave tile 32x64 (2x4 WMMA accums),
// block tile 128x128, K step 32, double-buffered LDS.
// A tile staged with CDNA5 async global->LDS (ASYNCcnt); B tile transposed
// through VGPRs so both fragments read as two ds_load_b128 per lane.
// act: 0=none 1=relu 2=elu
// ---------------------------------------------------------------------------
#define GT_M 128
#define GT_N 128
#define GT_K 32
#define LDSTR 40   // bf16 units; multiple of 8 -> keeps 16B alignment

__global__ __launch_bounds__(256) void k_gemm_bf16(
    const u16* __restrict__ A, const u16* __restrict__ B,
    const float* __restrict__ bias, float* __restrict__ C,
    int M, int K, int N, int act)
{
    __shared__ __align__(16) u16 lA[2][GT_M * LDSTR];
    __shared__ __align__(16) u16 lB[2][GT_N * LDSTR];   // transposed: [col][k]

    const int t     = threadIdx.x;
    const int wave  = t >> 5;
    const int lane  = t & 31;
    const int waveM = wave & 3;
    const int waveN = wave >> 2;
    const int bM    = blockIdx.y * GT_M;
    const int bN    = blockIdx.x * GT_N;
    const int lo16  = lane & 15;
    const int khalf = lane >> 4;

    v8f acc[2][4] = {};
    uint4 breg[2];

    // ---- async stage of A tile (128 rows x 32 k) into lA[buf] ----
    auto stage_A_async = [&](int k0, int buf) {
        #pragma unroll
        for (int i = t; i < 512; i += 256) {
            int row  = i >> 2;
            int part = (i & 3) * 8;
            int gr   = bM + row;
            if (gr < M) {
                const u16* gp = A + (size_t)gr * K + k0 + part;
                u16* lp = &lA[buf][row * LDSTR + part];
                asm volatile("global_load_async_to_lds_b128 %0, %1, off"
                             :: "v"((u32)(size_t)lp), "v"((const void*)gp)
                             : "memory");
            }
        }
    };
    // ---- B tile (32 k x 128 cols): global -> regs, regs -> transposed LDS ----
    auto load_B_regs = [&](int k0) {
        #pragma unroll
        for (int c = 0; c < 2; ++c) {
            int i = t + c * 256;
            int krow = i >> 4, cchunk = (i & 15) * 8;
            breg[c] = *(const uint4*)(B + (size_t)(k0 + krow) * N + bN + cchunk);
        }
    };
    auto store_B_lds = [&](int buf) {
        #pragma unroll
        for (int c = 0; c < 2; ++c) {
            int i = t + c * 256;
            int krow = i >> 4, cchunk = (i & 15) * 8;
            const u16* pv = (const u16*)&breg[c];
            #pragma unroll
            for (int j = 0; j < 8; ++j)
                lB[buf][(cchunk + j) * LDSTR + krow] = pv[j];
        }
    };

    // prologue: stage tile k=0
    stage_A_async(0, 0);
    load_B_regs(0);
    store_B_lds(0);
    asm volatile("s_wait_asynccnt 0x0" ::: "memory");
    __syncthreads();

    int buf = 0;
    for (int k0 = 0; k0 < K; k0 += GT_K) {
        const int nxt = buf ^ 1;
        const bool more = (k0 + GT_K) < K;
        if (more) {
            stage_A_async(k0 + GT_K, nxt);   // overlaps with WMMA below
            load_B_regs(k0 + GT_K);
        }

        // ---- fragments: two ds_load_b128 each (16-bit A layout, ISA 7.12.2) ----
        v16bf afr[2], bfr[4];
        #pragma unroll
        for (int mi = 0; mi < 2; ++mi) {
            int row = waveM * 32 + mi * 16 + lo16;
            union { uint4 q[2]; v16bf v; } u;
            u.q[0] = *(const uint4*)(&lA[buf][row * LDSTR + khalf * 8]);
            u.q[1] = *(const uint4*)(&lA[buf][row * LDSTR + 16 + khalf * 8]);
            afr[mi] = u.v;
        }
        #pragma unroll
        for (int ni = 0; ni < 4; ++ni) {
            int col = waveN * 64 + ni * 16 + lo16;
            union { uint4 q[2]; v16bf v; } u;
            u.q[0] = *(const uint4*)(&lB[buf][col * LDSTR + khalf * 8]);
            u.q[1] = *(const uint4*)(&lB[buf][col * LDSTR + 16 + khalf * 8]);
            bfr[ni] = u.v;
        }
        #pragma unroll
        for (int mi = 0; mi < 2; ++mi)
            #pragma unroll
            for (int ni = 0; ni < 4; ++ni)
                acc[mi][ni] = __builtin_amdgcn_wmma_f32_16x16x32_bf16(
                    false, afr[mi], false, bfr[ni], (short)0, acc[mi][ni], false, false);

        if (more) {
            store_B_lds(nxt);
            asm volatile("s_wait_asynccnt 0x0" ::: "memory");
        }
        __syncthreads();
        buf = nxt;
    }

    // ---- epilogue: bias + activation + store ----
    #pragma unroll
    for (int mi = 0; mi < 2; ++mi)
        #pragma unroll
        for (int ni = 0; ni < 4; ++ni) {
            int col = bN + waveN * 64 + ni * 16 + lo16;
            float bv = bias ? bias[col] : 0.0f;
            #pragma unroll
            for (int v = 0; v < 8; ++v) {
                int row = bM + waveM * 32 + mi * 16 + khalf * 8 + v;
                if (row < M) {
                    float x = acc[mi][ni][v] + bv;
                    if (act == 1) x = fmaxf(x, 0.0f);
                    else if (act == 2) x = (x > 0.0f) ? x : (expf(x) - 1.0f);
                    C[(size_t)row * N + col] = x;
                }
            }
        }
}

// ---------------------------------------------------------------------------
// GCN aggregation:  out = D^-1/2 (A+I) D^-1/2 H + b
// ---------------------------------------------------------------------------
__global__ void k_deg(const int* __restrict__ dst, float* __restrict__ deg, int E) {
    int e = blockIdx.x * blockDim.x + threadIdx.x;
    if (e < E) atomicAdd(&deg[dst[e]], 1.0f);
}

__global__ void k_gcn_self(const float* __restrict__ H, const float* __restrict__ dinv,
                           const float* __restrict__ bias, float* __restrict__ out,
                           int N, int C) {
    int i = blockIdx.x * blockDim.x + threadIdx.x;
    if (i < N * C) {
        int n = i / C, c = i - n * C;
        float di = dinv[n];
        out[i] = H[i] * di * di + bias[c];
    }
}

__global__ void k_gcn_edges(const float* __restrict__ H, const float* __restrict__ dinv,
                            const int* __restrict__ src, const int* __restrict__ dst,
                            float* __restrict__ out, int E, int C) {
    int idx = blockIdx.x * blockDim.x + threadIdx.x;
    int chunks = C >> 2;
    int e = idx / chunks;
    if (e < E) {
        int q = (idx - e * chunks) * 4;
        int s = src[e], d = dst[e];
        float nrm = dinv[s] * dinv[d];
        float4 hv = *(const float4*)(H + (size_t)s * C + q);
        float* op = out + (size_t)d * C + q;
        atomicAdd(op + 0, hv.x * nrm);
        atomicAdd(op + 1, hv.y * nrm);
        atomicAdd(op + 2, hv.z * nrm);
        atomicAdd(op + 3, hv.w * nrm);
    }
}

// ---------------------------------------------------------------------------
// GAT kernels
// ---------------------------------------------------------------------------
__device__ __forceinline__ u32 fenc(float f) {   // monotonic float->uint for atomic max
    u32 b = __float_as_uint(f);
    return (b & 0x80000000u) ? ~b : (b | 0x80000000u);
}
__device__ __forceinline__ float fdec(u32 k) {
    u32 b = (k & 0x80000000u) ? (k & 0x7FFFFFFFu) : ~k;
    return __uint_as_float(b);
}

__global__ void k_gat_scores(const float* __restrict__ Hf,
                             const float* __restrict__ a_src, const float* __restrict__ a_dst,
                             float* __restrict__ asrc, float* __restrict__ adst,
                             int N, int Hh, int C) {
    int i = blockIdx.x * blockDim.x + threadIdx.x;
    if (i < N * Hh) {
        int n = i / Hh, h = i - n * Hh;
        const float* hp = Hf + (size_t)n * Hh * C + (size_t)h * C;
        const float* as = a_src + h * C;
        const float* ad = a_dst + h * C;
        float s = 0.0f, d = 0.0f;
        for (int c = 0; c < C; ++c) { float v = hp[c]; s += v * as[c]; d += v * ad[c]; }
        asrc[i] = s; adst[i] = d;
    }
}

// edges 0..E-1 real, E..E+N-1 self loops
__global__ void k_gat_edge_logits(const float* __restrict__ asrc, const float* __restrict__ adst,
                                  const int* __restrict__ srcE, const int* __restrict__ dstE,
                                  float* __restrict__ elog, u32* __restrict__ emax_enc,
                                  int E, int N, int Hh) {
    int idx = blockIdx.x * blockDim.x + threadIdx.x;
    int tot = (E + N) * Hh;
    if (idx < tot) {
        int e = idx / Hh, h = idx - e * Hh;
        int s, d;
        if (e < E) { s = srcE[e]; d = dstE[e]; } else { s = d = e - E; }
        float v = asrc[s * Hh + h] + adst[d * Hh + h];
        v = (v >= 0.0f) ? v : 0.2f * v;                   // leaky_relu 0.2
        elog[idx] = v;
        atomicMax(&emax_enc[d * Hh + h], fenc(v));
    }
}

__global__ void k_gat_decode(const u32* __restrict__ enc, float* __restrict__ emax,
                             float* __restrict__ psum, int n) {
    int i = blockIdx.x * blockDim.x + threadIdx.x;
    if (i < n) { emax[i] = fdec(enc[i]); psum[i] = 0.0f; }
}

__global__ void k_gat_edge_exp(float* __restrict__ elog, const float* __restrict__ emax,
                               const int* __restrict__ srcE, const int* __restrict__ dstE,
                               float* __restrict__ psum, int E, int N, int Hh) {
    int idx = blockIdx.x * blockDim.x + threadIdx.x;
    int tot = (E + N) * Hh;
    if (idx < tot) {
        int e = idx / Hh, h = idx - e * Hh;
        int d = (e < E) ? dstE[e] : (e - E);
        float p = expf(elog[idx] - emax[d * Hh + h]);
        elog[idx] = p;
        atomicAdd(&psum[d * Hh + h], p);
    }
}

__global__ void k_gat_aggr(const float* __restrict__ Hf, const float* __restrict__ p,
                           const float* __restrict__ psum,
                           const int* __restrict__ srcE, const int* __restrict__ dstE,
                           float* __restrict__ out, int E, int N, int Hh, int C) {
    int idx = blockIdx.x * blockDim.x + threadIdx.x;
    int chunks = C >> 2;
    int per = Hh * chunks;
    int e = idx / per;
    if (e < E + N) {
        int r = idx - e * per;
        int h = r / chunks;
        int q = (r - h * chunks) * 4;
        int s, d;
        if (e < E) { s = srcE[e]; d = dstE[e]; } else { s = d = e - E; }
        float alpha = p[e * Hh + h] / (psum[d * Hh + h] + 1e-16f);
        float4 hv = *(const float4*)(Hf + (size_t)s * Hh * C + (size_t)h * C + q);
        float* op = out + (size_t)d * Hh * C + (size_t)h * C + q;
        atomicAdd(op + 0, hv.x * alpha);
        atomicAdd(op + 1, hv.y * alpha);
        atomicAdd(op + 2, hv.z * alpha);
        atomicAdd(op + 3, hv.w * alpha);
    }
}

__global__ void k_bias_elu(float* __restrict__ p, const float* __restrict__ bias, int N, int C) {
    int i = blockIdx.x * blockDim.x + threadIdx.x;
    if (i < N * C) {
        int c = i % C;
        float v = p[i] + bias[c];
        p[i] = (v > 0.0f) ? v : (expf(v) - 1.0f);
    }
}

// ---------------------------------------------------------------------------
// wave32 helpers + fused tail kernels
// ---------------------------------------------------------------------------
__device__ __forceinline__ float wave_sum(float v) {
    #pragma unroll
    for (int o = 16; o > 0; o >>= 1) v += __shfl_xor(v, o, 32);
    return v;
}
__device__ __forceinline__ float wave_max(float v) {
    #pragma unroll
    for (int o = 16; o > 0; o >>= 1) v = fmaxf(v, __shfl_xor(v, o, 32));
    return v;
}

// out = relu(log_softmax(in + bias)) over C=256, one wave per node
__global__ __launch_bounds__(256) void k_lsm_relu(const float* __restrict__ in,
                                                  const float* __restrict__ bias,
                                                  float* __restrict__ out, int N, int C) {
    int wave = threadIdx.x >> 5, lane = threadIdx.x & 31;
    int n = blockIdx.x * 8 + wave;
    if (n >= N) return;
    float v[8];
    float m = -INFINITY;
    #pragma unroll
    for (int t = 0; t < 8; ++t) {
        int c = lane + t * 32;
        v[t] = in[(size_t)n * C + c] + bias[c];
        m = fmaxf(m, v[t]);
    }
    m = wave_max(m);
    float s = 0.0f;
    #pragma unroll
    for (int t = 0; t < 8; ++t) s += expf(v[t] - m);
    s = wave_sum(s);
    float lse = m + logf(s);
    #pragma unroll
    for (int t = 0; t < 8; ++t) {
        int c = lane + t * 32;
        out[(size_t)n * C + c] = fmaxf(0.0f, v[t] - lse);
    }
}

// attention fusion over two 512-d views + MLP + log_softmax(7); one wave/node
__global__ __launch_bounds__(256) void k_fuse_final(
    const float* __restrict__ emb1, const float* __restrict__ emb3,
    const float* __restrict__ emb4, const float* __restrict__ emb6,
    const float* __restrict__ aw1, const float* __restrict__ ab1,
    const float* __restrict__ aw2,
    const float* __restrict__ mw1, const float* __restrict__ mb1,
    const float* __restrict__ mw2, const float* __restrict__ mb2,
    float* __restrict__ out, int N)
{
    int wave = threadIdx.x >> 5, lane = threadIdx.x & 31;
    int n = blockIdx.x * 8 + wave;
    if (n >= N) return;

    float zx[16], zy[16];
    #pragma unroll
    for (int t = 0; t < 16; ++t) {
        int f = lane + t * 32;
        zx[t] = (f < 256) ? emb1[(size_t)n * 256 + f] : emb3[(size_t)n * 256 + f - 256];
        zy[t] = (f < 256) ? emb4[(size_t)n * 256 + f] : emb6[(size_t)n * 256 + f - 256];
    }
    float w[2];
    for (int r = 0; r < 2; ++r) {
        const float* z = r ? zy : zx;
        float acc = 0.0f;
        for (int j = 0; j < 16; ++j) {
            float p = 0.0f;
            #pragma unroll
            for (int t = 0; t < 16; ++t) {
                int f = lane + t * 32;
                p += z[t] * aw1[f * 16 + j];
            }
            p = wave_sum(p);
            acc += tanhf(p + ab1[j]) * aw2[j];
        }
        w[r] = acc;
    }
    float mx = fmaxf(w[0], w[1]);
    float e0 = expf(w[0] - mx), e1 = expf(w[1] - mx);
    float inv = 1.0f / (e0 + e1);
    float b0 = e0 * inv, b1 = e1 * inv;

    float emb[16];
    #pragma unroll
    for (int t = 0; t < 16; ++t) emb[t] = b0 * zx[t] + b1 * zy[t];

    float h2[2] = {0.0f, 0.0f};
    for (int j = 0; j < 64; ++j) {
        float p = 0.0f;
        #pragma unroll
        for (int t = 0; t < 16; ++t) {
            int f = lane + t * 32;
            p += emb[t] * mw1[f * 64 + j];
        }
        p = wave_sum(p);
        float hv = tanhf(p + mb1[j]);
        if ((j & 31) == lane) h2[j >> 5] = hv;
    }
    float logit[7];
    for (int o = 0; o < 7; ++o) {
        float p = h2[0] * mw2[lane * 7 + o] + h2[1] * mw2[(lane + 32) * 7 + o];
        logit[o] = wave_sum(p) + mb2[o];
    }
    float m = logit[0];
    for (int o = 1; o < 7; ++o) m = fmaxf(m, logit[o]);
    float se = 0.0f;
    for (int o = 0; o < 7; ++o) se += expf(logit[o] - m);
    float lse = m + logf(se);
    if (lane < 7) out[(size_t)n * 7 + lane] = logit[lane] - lse;
}

// ---------------------------------------------------------------------------
// host orchestration
// ---------------------------------------------------------------------------
extern "C" void kernel_launch(void* const* d_in, const int* in_sizes, int n_in,
                              void* d_out, int out_size, void* d_ws, size_t ws_size,
                              hipStream_t stream) {
    const int NN = in_sizes[0] / 1024;          // 20000
    const int E  = in_sizes[1] / 2;             // 160000

    const float* x = (const float*)d_in[0];
    const int* sadj  = (const int*)d_in[1];
    const int* fadj  = (const int*)d_in[2];
    const int* asadj = (const int*)d_in[3];
    const int* afadj = (const int*)d_in[4];
    const float* W1a = (const float*)d_in[5];  const float* b1a = (const float*)d_in[6];
    const float* W1b = (const float*)d_in[7];  const float* b1b = (const float*)d_in[8];
    const float* W2a = (const float*)d_in[9];  const float* b2a = (const float*)d_in[10];
    const float* W2b = (const float*)d_in[11]; const float* b2b = (const float*)d_in[12];
    const float* W3a = (const float*)d_in[13]; const float* b3a = (const float*)d_in[14];
    const float* W3b = (const float*)d_in[15]; const float* b3b = (const float*)d_in[16];
    const float* W4a = (const float*)d_in[17]; const float* b4a = (const float*)d_in[18];
    const float* W4b = (const float*)d_in[19]; const float* b4b = (const float*)d_in[20];
    const float* Wg1a = (const float*)d_in[21]; const float* as1a = (const float*)d_in[22];
    const float* ad1a = (const float*)d_in[23]; const float* bg1a = (const float*)d_in[24];
    const float* Wg1b = (const float*)d_in[25]; const float* as1b = (const float*)d_in[26];
    const float* ad1b = (const float*)d_in[27]; const float* bg1b = (const float*)d_in[28];
    const float* Wg2a = (const float*)d_in[29]; const float* as2a = (const float*)d_in[30];
    const float* ad2a = (const float*)d_in[31]; const float* bg2a = (const float*)d_in[32];
    const float* Wg2b = (const float*)d_in[33]; const float* as2b = (const float*)d_in[34];
    const float* ad2b = (const float*)d_in[35]; const float* bg2b = (const float*)d_in[36];
    const float* aw1 = (const float*)d_in[37]; const float* ab1 = (const float*)d_in[38];
    const float* aw2 = (const float*)d_in[39];
    const float* mw1 = (const float*)d_in[40]; const float* mb1 = (const float*)d_in[41];
    const float* mw2 = (const float*)d_in[42]; const float* mb2 = (const float*)d_in[43];

    // ---- workspace carve ----
    char* wp = (char*)d_ws;
    auto carve = [&](size_t bytes) -> void* {
        void* p = (void*)wp;
        wp += (bytes + 255) & ~(size_t)255;
        return p;
    };
    u16* W1a_bf  = (u16*)carve((size_t)1024*512*2);
    u16* W1b_bf  = (u16*)carve((size_t)512*256*2);
    u16* W2a_bf  = (u16*)carve((size_t)1024*512*2);
    u16* W2b_bf  = (u16*)carve((size_t)512*256*2);
    u16* W3a_bf  = (u16*)carve((size_t)256*128*2);
    u16* W3b_bf  = (u16*)carve((size_t)128*256*2);
    u16* W4a_bf  = (u16*)carve((size_t)256*128*2);
    u16* W4b_bf  = (u16*)carve((size_t)128*256*2);
    u16* Wg1a_bf = (u16*)carve((size_t)256*512*2);
    u16* Wg1b_bf = (u16*)carve((size_t)512*256*2);
    u16* Wg2a_bf = (u16*)carve((size_t)256*512*2);
    u16* Wg2b_bf = (u16*)carve((size_t)512*256*2);
    u16*   bfA   = (u16*)carve((size_t)NN*1024*2);
    float* P     = (float*)carve((size_t)NN*512*4);
    float* Q     = (float*)carve((size_t)NN*512*4);
    float* R     = (float*)carve((size_t)NN*256*4);
    float* emb1  = (float*)carve((size_t)NN*256*4);
    float* emb3  = (float*)carve((size_t)NN*256*4);
    float* emb4  = (float*)carve((size_t)NN*256*4);
    float* emb6  = (float*)carve((size_t)NN*256*4);
    float* dinv_s = (float*)carve((size_t)NN*4);
    float* dinv_f = (float*)carve((size_t)NN*4);
    float* asrcB = (float*)carve((size_t)NN*4*4);
    float* adstB = (float*)carve((size_t)NN*4*4);
    float* emaxf = (float*)carve((size_t)NN*4*4);
    float* psum  = (float*)carve((size_t)NN*4*4);
    u32*   emax_enc = (u32*)carve((size_t)NN*4*4);
    float* elog  = (float*)carve((size_t)(E+NN)*4*4);

    auto grid1 = [](int n) { return dim3((unsigned)((n + 255) / 256)); };
    auto cast  = [&](const float* s, u16* d, int n) {
        k_cast_bf16<<<grid1(n), 256, 0, stream>>>(s, d, n);
    };
    auto gemm = [&](const u16* A, const u16* B, const float* bias, float* C,
                    int M, int K, int Nn, int act) {
        dim3 g((unsigned)(Nn / 128), (unsigned)((M + 127) / 128));
        k_gemm_bf16<<<g, 256, 0, stream>>>(A, B, bias, C, M, K, Nn, act);
    };
    auto fill = [&](float* p, float v, int n) {
        k_fill_f32<<<grid1(n), 256, 0, stream>>>(p, v, n);
    };

    // ---- weight conversion (once per call) ----
    cast(W1a, W1a_bf, 1024*512);  cast(W1b, W1b_bf, 512*256);
    cast(W2a, W2a_bf, 1024*512);  cast(W2b, W2b_bf, 512*256);
    cast(W3a, W3a_bf, 256*128);   cast(W3b, W3b_bf, 128*256);
    cast(W4a, W4a_bf, 256*128);   cast(W4b, W4b_bf, 128*256);
    cast(Wg1a, Wg1a_bf, 256*512); cast(Wg1b, Wg1b_bf, 512*256);
    cast(Wg2a, Wg2a_bf, 256*512); cast(Wg2b, Wg2b_bf, 512*256);

    // ---- GCN normalization factors (deg init 1.0 covers self loop) ----
    fill(dinv_s, 1.0f, NN);
    k_deg<<<grid1(E), 256, 0, stream>>>(sadj + E, dinv_s, E);
    k_rsqrt<<<grid1(NN), 256, 0, stream>>>(dinv_s, NN);
    fill(dinv_f, 1.0f, NN);
    k_deg<<<grid1(E), 256, 0, stream>>>(fadj + E, dinv_f, E);
    k_rsqrt<<<grid1(NN), 256, 0, stream>>>(dinv_f, NN);

    // ---- GCN block: in[NN,K1] -> out[NN,N2] (relu'd) ----
    auto gcn_block = [&](const float* in, int K1,
                         const u16* Wa, const float* ba, int N1,
                         const u16* Wb, const float* bb, int N2,
                         const float* dinv, const int* srcp, const int* dstp,
                         float* outp) {
        cast(in, bfA, NN * K1);
        gemm(bfA, Wa, nullptr, P, NN, K1, N1, 0);
        k_gcn_self<<<grid1(NN*N1), 256, 0, stream>>>(P, dinv, ba, Q, NN, N1);
        k_gcn_edges<<<grid1(E*(N1/4)), 256, 0, stream>>>(P, dinv, srcp, dstp, Q, E, N1);
        k_relu<<<grid1(NN*N1), 256, 0, stream>>>(Q, NN*N1);
        cast(Q, bfA, NN * N1);
        gemm(bfA, Wb, nullptr, P, NN, N1, N2, 0);
        k_gcn_self<<<grid1(NN*N2), 256, 0, stream>>>(P, dinv, bb, outp, NN, N2);
        k_gcn_edges<<<grid1(E*(N2/4)), 256, 0, stream>>>(P, dinv, srcp, dstp, outp, E, N2);
        k_relu<<<grid1(NN*N2), 256, 0, stream>>>(outp, NN*N2);
    };

    // ---- one GAT layer: hfeat = in@W ; softmax-attention aggregate -> outp ----
    auto gat_layer = [&](const float* in, int K, const u16* Wbf, int Hh, int C,
                         const float* a_s, const float* a_d,
                         const int* srcp, const int* dstp, float* outp) {
        int Etot = E + NN;
        cast(in, bfA, NN * K);
        gemm(bfA, Wbf, nullptr, P, NN, K, Hh * C, 0);
        k_gat_scores<<<grid1(NN*Hh), 256, 0, stream>>>(P, a_s, a_d, asrcB, adstB, NN, Hh, C);
        fill((float*)emax_enc, 0.0f, NN * Hh);   // encoded 0 < enc of any float: identity
        k_gat_edge_logits<<<grid1(Etot*Hh), 256, 0, stream>>>(asrcB, adstB, srcp, dstp,
                                                              elog, emax_enc, E, NN, Hh);
        k_gat_decode<<<grid1(NN*Hh), 256, 0, stream>>>(emax_enc, emaxf, psum, NN*Hh);
        k_gat_edge_exp<<<grid1(Etot*Hh), 256, 0, stream>>>(elog, emaxf, srcp, dstp,
                                                           psum, E, NN, Hh);
        fill(outp, 0.0f, NN * Hh * C);
        k_gat_aggr<<<grid1(Etot*Hh*(C/4)), 256, 0, stream>>>(P, elog, psum, srcp, dstp,
                                                             outp, E, NN, Hh, C);
    };

    // ---- GAT block: in[NN,256] -> relu(log_softmax(gat_b(elu(gat_a)))) -> outR[NN,256]
    auto gat_block = [&](const float* in, const int* srcp, const int* dstp,
                         const u16* Wa, const float* asa, const float* ada, const float* bga,
                         const u16* Wb, const float* asb, const float* adb, const float* bgb,
                         float* outR) {
        gat_layer(in, 256, Wa, 4, 128, asa, ada, srcp, dstp, Q);
        k_bias_elu<<<grid1(NN*512), 256, 0, stream>>>(Q, bga, NN, 512);
        gat_layer(Q, 512, Wb, 1, 256, asb, adb, srcp, dstp, Q);   // cast(Q) happens first
        k_lsm_relu<<<dim3((unsigned)((NN+7)/8)), 256, 0, stream>>>(Q, bgb, outR, NN, 256);
    };

    // ---- branch 1 (sadj / asadj) ----
    gcn_block(x, 1024, W1a_bf, b1a, 512, W1b_bf, b1b, 256, dinv_s, sadj, sadj + E, emb1);
    gat_block(emb1, asadj, asadj + E, Wg1a_bf, as1a, ad1a, bg1a,
              Wg1b_bf, as1b, ad1b, bg1b, R);                       // emb2 in R
    gcn_block(R, 256, W3a_bf, b3a, 128, W3b_bf, b3b, 256, dinv_s, sadj, sadj + E, emb3);

    // ---- branch 2 (fadj / afadj) ----
    gcn_block(x, 1024, W2a_bf, b2a, 512, W2b_bf, b2b, 256, dinv_f, fadj, fadj + E, emb4);
    gat_block(emb4, afadj, afadj + E, Wg2a_bf, as2a, ad2a, bg2a,
              Wg2b_bf, as2b, ad2b, bg2b, R);                       // emb5 in R
    gcn_block(R, 256, W4a_bf, b4a, 128, W4b_bf, b4b, 256, dinv_f, fadj, fadj + E, emb6);

    // ---- attention fusion + MLP + log_softmax ----
    k_fuse_final<<<dim3((unsigned)((NN+7)/8)), 256, 0, stream>>>(
        emb1, emb3, emb4, emb6, aw1, ab1, aw2, mw1, mb1, mw2, mb2,
        (float*)d_out, NN);

    (void)n_in; (void)out_size; (void)ws_size;
}
